// cmaBlock_89670327206182
// MI455X (gfx1250) — matrix operand order repaired and checked by reading, not compiled, not run on hardware
//
#include <hip/hip_runtime.h>

// cmaBlock for MI455X (gfx1250), wave32 + v_wmma_f32_16x16x32_f16.
// B=8, C=256, H=W=112.
// Roofline: ~64 GFLOP total, min HBM traffic ~0.8 GB with f16 intermediates.
// Design: full-K (256) activation tiles staged once per block in LDS, M-loop
// inside the kernel so every activation byte is read from HBM exactly once.
// Attention (QK -> softmax -> MV) fully fused per (b,c) slice in LDS.

typedef __attribute__((ext_vector_type(16))) _Float16 v16h;
typedef __attribute__((ext_vector_type(8)))  float    v8f;

#define BDIM   8
#define CDIM   256
#define NPIX   (112 * 112)   // 12544

union FragU { v16h v; unsigned int u[8]; };

__device__ __forceinline__ unsigned int lds_pair(const _Float16* p) {
    unsigned int v;
    __builtin_memcpy(&v, p, 4);   // two contiguous f16 -> one b32 LDS load
    return v;
}

// A-fragment K offset for element pair j (ISA 16-bit A 16x32 layout):
// lanes 0-15 hold K {0..7,16..23}, lanes 16-31 hold K {8..15,24..31}.
__device__ __forceinline__ int a_kb(int j, int half) {
    return (j < 4) ? (2 * j + 8 * half) : (8 + 2 * j + 8 * half);
}
// B-fragment (32x16, staged column-major): lanes 0-15 K {0..7,16..23}, 16-31 K {8..15,24..31}.
__device__ __forceinline__ int b_kb(int j, int half) {
    return (j < 4) ? (2 * j + 8 * half) : (8 + 2 * j + 8 * half);
}

// ---------------------------------------------------------------------------
// Projection GEMM: Out[b, m, p] = sum_k Wstk[m, k] * X[b, k, p], f32 -> f16.
// Wstk is the stacked weight [W0 ; W1] (M = mtiles*128). One block owns a
// 128-pixel column strip; X tile (256K x 128N, f16, 64KB) staged ONCE, then
// M-loop stages 128x256 weight tiles. 8 waves = 4(M) x 2(N); wave = 32x64.
// ---------------------------------------------------------------------------
__global__ void __launch_bounds__(256) proj_f32_to_f16(
    const float* __restrict__ W0,    // rows 0..255 of stacked weights
    const float* __restrict__ W1,    // rows 256..511 (unused when mtiles==2)
    const float* __restrict__ X,     // [B, 256, NPIX] f32
    _Float16*    __restrict__ O0,    // [B, 256, NPIX] f16
    _Float16*    __restrict__ O1,    // second output (unused when mtiles==2)
    int mtiles)
{
    extern __shared__ char smem[];
    _Float16* sX = (_Float16*)smem;                  // [n=128][k=256] col-major
    _Float16* sW = (_Float16*)(smem + 128 * 256 * 2); // [m=128][k=256] row-major

    const int tid  = threadIdx.x;
    const int lane = tid & 31;
    const int wv   = tid >> 5;
    const int half = lane >> 4;
    const int l16  = lane & 15;
    const int n0   = blockIdx.x * 128;
    const int b    = blockIdx.z;
    const float* Xb = X + (size_t)b * CDIM * NPIX;

    const int wm = wv >> 1;   // 0..3 -> 32-row group
    const int wn = wv & 1;    // 0..1 -> 64-col group

    // Stage full-K activation tile once (coalesced along n).
    for (int idx = tid; idx < 256 * 128; idx += 256) {
        int k = idx >> 7, n = idx & 127;
        sX[n * 256 + k] = (_Float16)Xb[(size_t)k * NPIX + n0 + n];
    }

    for (int mt = 0; mt < mtiles; ++mt) {
        const int mg0 = mt * 128;
        const float* Wsrc = (mg0 < 256) ? W0 : W1;
        _Float16*    Odst = (mg0 < 256) ? O0 : O1;
        const int row0 = (mg0 < 256) ? mg0 : (mg0 - 256);

        __syncthreads();   // sX ready (mt==0) / previous readers done with sW
        for (int idx = tid; idx < 128 * 256; idx += 256) {
            int m = idx >> 8, k = idx & 255;
            sW[idx] = (_Float16)Wsrc[(row0 + m) * CDIM + k];
        }
        __syncthreads();

        v8f acc[2][4] = {};
        for (int kk = 0; kk < 256; kk += 32) {
            FragU a[2], bf[4];
            #pragma unroll
            for (int t = 0; t < 2; ++t) {
                int row = wm * 32 + t * 16 + l16;
                #pragma unroll
                for (int j = 0; j < 8; ++j)
                    a[t].u[j] = lds_pair(&sW[row * 256 + kk + a_kb(j, half)]);
            }
            #pragma unroll
            for (int t = 0; t < 4; ++t) {
                int col = wn * 64 + t * 16 + l16;
                #pragma unroll
                for (int j = 0; j < 8; ++j)
                    bf[t].u[j] = lds_pair(&sX[col * 256 + kk + b_kb(j, half)]);
            }
            #pragma unroll
            for (int tm = 0; tm < 2; ++tm)
                #pragma unroll
                for (int tn = 0; tn < 4; ++tn)
                    acc[tm][tn] = __builtin_amdgcn_wmma_f32_16x16x32_f16(
                        false, a[tm].v, false, bf[tn].v, (short)0,
                        acc[tm][tn], false, false);
        }

        // D layout: lane n = l16, VGPR r -> M = r + 8*half.
        #pragma unroll
        for (int tm = 0; tm < 2; ++tm)
            #pragma unroll
            for (int tn = 0; tn < 4; ++tn) {
                int o = row0 + wm * 32 + tm * 16 + 8 * half;
                int p = n0 + wn * 64 + tn * 16 + l16;
                _Float16* dst = Odst + ((size_t)b * CDIM + o) * NPIX + p;
                #pragma unroll
                for (int r = 0; r < 8; ++r)
                    dst[(size_t)r * NPIX] = (_Float16)acc[tm][tn][r];
            }
    }
}

// ---------------------------------------------------------------------------
// Attention per (b,c) slice: S = Q@K (112x112), row softmax, Z = M@V.
// Slice lives in LDS (~105 KB); K-dim padded 112 -> 128 with zeros.
// Q is staged with CDNA5 async global->LDS copies (ASYNCcnt); K/V require a
// transpose so they go through VGPRs with coalesced global reads.
// ---------------------------------------------------------------------------
__global__ void __launch_bounds__(256) attn_slice(
    const _Float16* __restrict__ Q,
    const _Float16* __restrict__ Km,
    const _Float16* __restrict__ V,
    _Float16*       __restrict__ Zout)
{
    extern __shared__ char smem[];
    _Float16* sA = (_Float16*)smem;        // [112][128] row-major: Q, then M
    _Float16* sB = sA + 112 * 128;         // [112 cols][128 k] col-major: K, then V
    float*    sS = (float*)(smem + 2 * 112 * 128 * 2); // [112][112] scores

    const int c = blockIdx.x, b = blockIdx.y;
    const size_t base = ((size_t)b * CDIM + c) * NPIX;
    const int tid  = threadIdx.x;
    const int lane = tid & 31;
    const int wv   = tid >> 5;
    const int half = lane >> 4;
    const int l16  = lane & 15;

    // Zero the K-dim padding (cols 112..127 of sA rows / rows of sB columns).
    for (int idx = tid; idx < 112 * 16; idx += 256) {
        int i = idx >> 4, t = 112 + (idx & 15);
        sA[i * 128 + t] = (_Float16)0.0f;
        sB[i * 128 + t] = (_Float16)0.0f;
    }
    // Q: layout-preserving copy -> async DMA to LDS, 16B per lane per op.
    // 112 rows x 14 16-byte chunks per row.
    for (int idx = tid; idx < 112 * 14; idx += 256) {
        int i = idx / 14, cch = idx % 14;
        unsigned lds = (unsigned)(unsigned long long)(sA + i * 128 + cch * 8);
        unsigned long long ga =
            (unsigned long long)(Q + base + (size_t)i * 112 + cch * 8);
        asm volatile("global_load_async_to_lds_b128 %0, %1, off"
                     :: "v"(lds), "v"(ga) : "memory");
    }
    // K: coalesced global read, transposed LDS write sB[j][t] = K[t][j].
    for (int idx = tid; idx < NPIX; idx += 256) {
        int t = idx / 112, j = idx % 112;
        sB[j * 128 + t] = Km[base + idx];
    }
    asm volatile("s_wait_asynccnt 0x0" ::: "memory");
    __syncthreads();

    // Phase 2: S = Q @ K over 7x7 = 49 tiles, K-loop 4 x 32.
    for (int ti = wv; ti < 49; ti += 8) {
        int tm = ti / 7, tn = ti % 7;
        v8f acc = {};
        for (int kk = 0; kk < 128; kk += 32) {
            FragU a, bf;
            int row = tm * 16 + l16;
            int col = tn * 16 + l16;
            #pragma unroll
            for (int j = 0; j < 8; ++j) {
                a.u[j]  = lds_pair(&sA[row * 128 + kk + a_kb(j, half)]);
                bf.u[j] = lds_pair(&sB[col * 128 + kk + b_kb(j, half)]);
            }
            acc = __builtin_amdgcn_wmma_f32_16x16x32_f16(
                false, a.v, false, bf.v, (short)0, acc, false, false);
        }
        #pragma unroll
        for (int r = 0; r < 8; ++r)
            sS[(tm * 16 + r + 8 * half) * 112 + tn * 16 + l16] = acc[r];
    }
    __syncthreads();

    // Phase 3: V -> sB (transposed; pad rows still zero); softmax -> sA (f16).
    for (int idx = tid; idx < NPIX; idx += 256) {
        int t = idx / 112, j = idx % 112;
        sB[j * 128 + t] = V[base + idx];
    }
    if (tid < 112) {
        int i = tid;
        float mx = -3.4e38f;
        for (int j = 0; j < 112; ++j) mx = fmaxf(mx, sS[i * 112 + j]);
        float s = 0.0f;
        for (int j = 0; j < 112; ++j) {
            float e = __expf(sS[i * 112 + j] - mx);
            sS[i * 112 + j] = e;
            s += e;
        }
        float inv = 1.0f / s;
        for (int j = 0; j < 112; ++j)
            sA[i * 128 + j] = (_Float16)(sS[i * 112 + j] * inv); // pads stay 0
    }
    __syncthreads();

    // Phase 4: Z = M @ V -> global f16.
    for (int ti = wv; ti < 49; ti += 8) {
        int tm = ti / 7, tn = ti % 7;
        v8f acc = {};
        for (int kk = 0; kk < 128; kk += 32) {
            FragU a, bf;
            int row = tm * 16 + l16;
            int col = tn * 16 + l16;
            #pragma unroll
            for (int j = 0; j < 8; ++j) {
                a.u[j]  = lds_pair(&sA[row * 128 + kk + a_kb(j, half)]);
                bf.u[j] = lds_pair(&sB[col * 128 + kk + b_kb(j, half)]);
            }
            acc = __builtin_amdgcn_wmma_f32_16x16x32_f16(
                false, a.v, false, bf.v, (short)0, acc, false, false);
        }
        #pragma unroll
        for (int r = 0; r < 8; ++r)
            Zout[base + (tm * 16 + r + 8 * half) * 112 + tn * 16 + l16] =
                (_Float16)acc[r];
    }
}

// ---------------------------------------------------------------------------
// Final conv + residual: out[b,o,p] = relu(x[b,o,p] + sum_c WZ[o,c]*Z[b,c,p]).
// Same full-K staged structure; f16 B-operand; fp32 epilogue.
// ---------------------------------------------------------------------------
__global__ void __launch_bounds__(256) convZ_residual_relu(
    const float*    __restrict__ Wmat,   // WZ [256,256]
    const _Float16* __restrict__ Zin,    // [B,256,NPIX] f16
    const float*    __restrict__ Xres,   // x [B,256,NPIX] f32
    float*          __restrict__ Out)    // [B,256,NPIX] f32
{
    extern __shared__ char smem[];
    _Float16* sX = (_Float16*)smem;                   // [n=128][k=256]
    _Float16* sW = (_Float16*)(smem + 128 * 256 * 2); // [m=128][k=256]

    const int tid  = threadIdx.x;
    const int lane = tid & 31;
    const int wv   = tid >> 5;
    const int half = lane >> 4;
    const int l16  = lane & 15;
    const int n0   = blockIdx.x * 128;
    const int b    = blockIdx.z;
    const _Float16* Zb = Zin + (size_t)b * CDIM * NPIX;

    const int wm = wv >> 1;
    const int wn = wv & 1;

    for (int idx = tid; idx < 256 * 128; idx += 256) {
        int k = idx >> 7, n = idx & 127;
        sX[n * 256 + k] = Zb[(size_t)k * NPIX + n0 + n];
    }

    for (int mt = 0; mt < 2; ++mt) {
        const int row0 = mt * 128;
        __syncthreads();
        for (int idx = tid; idx < 128 * 256; idx += 256) {
            int m = idx >> 8, k = idx & 255;
            sW[idx] = (_Float16)Wmat[(row0 + m) * CDIM + k];
        }
        __syncthreads();

        v8f acc[2][4] = {};
        for (int kk = 0; kk < 256; kk += 32) {
            FragU a[2], bf[4];
            #pragma unroll
            for (int t = 0; t < 2; ++t) {
                int row = wm * 32 + t * 16 + l16;
                #pragma unroll
                for (int j = 0; j < 8; ++j)
                    a[t].u[j] = lds_pair(&sW[row * 256 + kk + a_kb(j, half)]);
            }
            #pragma unroll
            for (int t = 0; t < 4; ++t) {
                int col = wn * 64 + t * 16 + l16;
                #pragma unroll
                for (int j = 0; j < 8; ++j)
                    bf[t].u[j] = lds_pair(&sX[col * 256 + kk + b_kb(j, half)]);
            }
            #pragma unroll
            for (int tm = 0; tm < 2; ++tm)
                #pragma unroll
                for (int tn = 0; tn < 4; ++tn)
                    acc[tm][tn] = __builtin_amdgcn_wmma_f32_16x16x32_f16(
                        false, a[tm].v, false, bf[tn].v, (short)0,
                        acc[tm][tn], false, false);
        }

        #pragma unroll
        for (int tm = 0; tm < 2; ++tm)
            #pragma unroll
            for (int tn = 0; tn < 4; ++tn) {
                int o = row0 + wm * 32 + tm * 16 + 8 * half;
                int p = n0 + wn * 64 + tn * 16 + l16;
                size_t off = ((size_t)b * CDIM + o) * NPIX + p;
                #pragma unroll
                for (int r = 0; r < 8; ++r) {
                    float v = Xres[off + (size_t)r * NPIX] + acc[tm][tn][r];
                    Out[off + (size_t)r * NPIX] = fmaxf(v, 0.0f);
                }
            }
    }
}

// ---------------------------------------------------------------------------
extern "C" void kernel_launch(void* const* d_in, const int* in_sizes, int n_in,
                              void* d_out, int out_size, void* d_ws, size_t ws_size,
                              hipStream_t stream) {
    const float* x  = (const float*)d_in[0];
    const float* y  = (const float*)d_in[1];
    const float* WQ = (const float*)d_in[2];
    const float* WK = (const float*)d_in[3];
    const float* WV = (const float*)d_in[4];
    const float* WZ = (const float*)d_in[5];
    float* out = (float*)d_out;

    const size_t elems = (size_t)BDIM * CDIM * NPIX;   // 25,690,112
    char* ws = (char*)d_ws;
    _Float16* Qb = (_Float16*)(ws);
    _Float16* Kb = (_Float16*)(ws + 2 * elems);
    _Float16* Vb = (_Float16*)(ws + 4 * elems);
    _Float16* Zb = (_Float16*)(ws + 6 * elems);        // ~196 MB total

    dim3 block(256);
    dim3 gridP(NPIX / 128, 1, BDIM);                   // 98 x 1 x 8
    size_t ldsP = (size_t)(2 * 128 * 256 * 2);         // 131,072 B

    // Q projection (x read once), fused K+V projection (y read once).
    proj_f32_to_f16<<<gridP, block, ldsP, stream>>>(WQ, WQ, x, Qb, Qb, 2);
    proj_f32_to_f16<<<gridP, block, ldsP, stream>>>(WK, WV, y, Kb, Vb, 4);

    dim3 gridA(CDIM, BDIM);                            // 2048 slices
    size_t ldsA = (size_t)(2 * 112 * 128 * 2) + (size_t)(112 * 112 * 4);
    attn_slice<<<gridA, block, ldsA, stream>>>(Qb, Kb, Vb, Zb);

    convZ_residual_relu<<<gridP, block, ldsP, stream>>>(WZ, Zb, x, out);
}
